// SASRec_10866267259552
// MI455X (gfx1250) — matrix operand order repaired and checked by reading, not compiled, not run on hardware
//
#include <hip/hip_runtime.h>

typedef __attribute__((ext_vector_type(16))) _Float16 v16h;
typedef __attribute__((ext_vector_type(8)))  _Float16 v8h;
typedef __attribute__((ext_vector_type(8)))  float    v8f;
typedef __attribute__((ext_vector_type(4)))  unsigned int u32x4;
typedef __attribute__((ext_vector_type(8)))  int      i32x8;
typedef __attribute__((ext_vector_type(4)))  int      i32x4;
typedef _Float16 half_t;

#define WMMA_F32_F16(a, b, c) \
  __builtin_amdgcn_wmma_f32_16x16x32_f16(false, (a), false, (b), (short)0, (c), false, false)

static constexpr int Bsz = 512, Ssz = 200, Dsz = 256, Hh = 4, DHd = 64, NBk = 2;
static constexpr int Mtok = Bsz * Ssz;   // 102400 tokens
static constexpr float EPS = 1e-8f;

// ---------------------------------------------------------------------------
// WMMA fragment loaders (wave32; A/B in this code's contiguous k order, C/D per CDNA5 ISA §7.12.2)
// A 16x32 f16: lane m=lane&15; halves 0-15 -> K=(lane/16)*16+0..15 (contig)
// B 32x16 f16: lane n=lane&15; halves 0-15 -> K=(lane/16)*16+0..15 (contig)
// C/D 16x16 f32: lane n=lane&15; vgpr r -> m=(lane/16)*8+r
// ---------------------------------------------------------------------------
__device__ __forceinline__ v16h load_a_frag(const half_t* base, int ld) {
  int lane = threadIdx.x & 31;
  int m = lane & 15;
  int k0 = (lane >> 4) * 16;
  const half_t* p = base + (size_t)m * ld + k0;
  v8h lo = *(const v8h*)(p);
  v8h hi = *(const v8h*)(p + 8);
  v16h r;
#pragma unroll
  for (int i = 0; i < 8; i++) { r[i] = lo[i]; r[i + 8] = hi[i]; }
  return r;
}

__device__ __forceinline__ v16h load_b_frag(const half_t* base, int ld) {
  int lane = threadIdx.x & 31;
  int n = lane & 15;
  int k0 = (lane >> 4) * 16;
  return *(const v16h*)(base + (size_t)n * ld + k0);
}

// q rows (clamped) from qkv buffer [M,768], cols [0,256): head h, d-chunk c
__device__ __forceinline__ v16h load_a_q(const half_t* qkvH, int b, int qb, int h, int c) {
  int lane = threadIdx.x & 31;
  int m = lane & 15;
  int qi = qb * 16 + m; if (qi > Ssz - 1) qi = Ssz - 1;
  const half_t* p = qkvH + (size_t)(b * Ssz + qi) * 768 + h * DHd + c * 32 + ((lane >> 4) * 16);
  v8h lo = *(const v8h*)(p);
  v8h hi = *(const v8h*)(p + 8);
  v16h r;
#pragma unroll
  for (int i = 0; i < 8; i++) { r[i] = lo[i]; r[i + 8] = hi[i]; }
  return r;
}

// k rows (clamped) as B fragment: B[d, kj] = k[kj, d]; lane n = key index
__device__ __forceinline__ v16h load_b_k(const half_t* qkvH, int b, int kt, int h, int c) {
  int lane = threadIdx.x & 31;
  int n = lane & 15;
  int kj = kt * 16 + n; if (kj > Ssz - 1) kj = Ssz - 1;
  const half_t* p = qkvH + (size_t)(b * Ssz + kj) * 768 + 256 + h * DHd + c * 32 + ((lane >> 4) * 16);
  return *(const v16h*)p;
}

// ---------------------------------------------------------------------------
// Tensor Data Mover: 2D tile (rows x cols of f16) from global -> LDS.
// Descriptor layout per CDNA5 ISA (D# groups 0/1; 2D so
// groups 2/3 are zero). data_size=1 (2 bytes). lds byte address = low 32 bits
// of the generic shared pointer (LDS aperture truncation).
// ---------------------------------------------------------------------------
#if __has_builtin(__builtin_amdgcn_tensor_load_to_lds)
#define HAVE_TDM 1
__device__ __forceinline__ void tdm_load_2d_f16(const half_t* gsrc, unsigned lds_addr,
                                                unsigned rows, unsigned cols,
                                                unsigned rowStrideElems) {
  unsigned long long ga = (unsigned long long)(size_t)gsrc;
  u32x4 g0;
  g0[0] = 1u;                                    // count=1 (valid user descriptor)
  g0[1] = lds_addr;                              // lds_addr [63:32]
  g0[2] = (unsigned)(ga & 0xFFFFFFFFull);        // global_addr [95:64]
  g0[3] = (unsigned)((ga >> 32) & 0x1FFFFFFull)  // global_addr [120:96]
        | (2u << 30);                            // type=2 ("image")
  i32x8 g1;
  g1[0] = (int)(1u << 16);                       // data_size=1 (2B); wg_mask=0
  g1[1] = (int)((cols & 0xFFFFu) << 16);         // tensor_dim0[15:0] at bits[63:48]
  g1[2] = (int)((cols >> 16) | ((rows & 0xFFFFu) << 16));  // dim0 hi | dim1 lo
  g1[3] = (int)((rows >> 16) | ((cols & 0xFFFFu) << 16));  // dim1 hi | tile_dim0
  g1[4] = (int)(rows & 0xFFFFu);                 // tile_dim1 (tile_dim2=0)
  g1[5] = (int)rowStrideElems;                   // tensor_dim0_stride[31:0]
  g1[6] = 0;                                     // dim0_stride hi | dim1_stride lo
  g1[7] = 0;                                     // tensor_dim1_stride hi
  i32x4 g2 = {0, 0, 0, 0};
  i32x4 g3 = {0, 0, 0, 0};
#if __clang_major__ >= 23
  i32x8 g4 = {0, 0, 0, 0, 0, 0, 0, 0};
  __builtin_amdgcn_tensor_load_to_lds(g0, g1, g2, g3, g4, 0);
#else
  __builtin_amdgcn_tensor_load_to_lds(g0, g1, g2, g3, 0);
#endif
}
#else
#define HAVE_TDM 0
#endif

// ---------------------------------------------------------------------------
__global__ void k_cvt(const float* __restrict__ src, half_t* __restrict__ dst, int n) {
  int i = blockIdx.x * blockDim.x + threadIdx.x;
  if (i < n) dst[i] = (half_t)src[i];
}

__global__ void k_embed(const float* __restrict__ item_emb, const float* __restrict__ pos_emb,
                        const int* __restrict__ log_seqs,
                        float* __restrict__ seqsF, half_t* __restrict__ seqsH,
                        float* __restrict__ maskF) {
  int t = blockIdx.x, d = threadIdx.x;
  int it = log_seqs[t];
  float msk = (it != 0) ? 1.f : 0.f;
  int s = t % Ssz;
  float v = (item_emb[(size_t)it * Dsz + d] * 16.0f + pos_emb[(size_t)s * Dsz + d]) * msk;
  seqsF[(size_t)t * Dsz + d] = v;
  seqsH[(size_t)t * Dsz + d] = (half_t)v;
  if (d == 0) maskF[t] = msk;
}

__device__ __forceinline__ float block_reduce(float v, float* red) {
  int d = threadIdx.x;
  red[d] = v; __syncthreads();
#pragma unroll
  for (int o = 128; o > 0; o >>= 1) { if (d < o) red[d] += red[d + o]; __syncthreads(); }
  float r = red[0]; __syncthreads();
  return r;
}

__global__ void k_layernorm(const float* __restrict__ X, const float* __restrict__ g,
                            const float* __restrict__ bb,
                            float* __restrict__ outF, half_t* __restrict__ outH) {
  __shared__ float red[256];
  int t = blockIdx.x, d = threadIdx.x;
  float x = X[(size_t)t * Dsz + d];
  float mean = block_reduce(x, red) * (1.0f / Dsz);
  float c = x - mean;
  float var = block_reduce(c * c, red) * (1.0f / Dsz);
  float y = c * rsqrtf(var + EPS) * g[d] + bb[d];
  outF[(size_t)t * Dsz + d] = y;
  if (outH) outH[(size_t)t * Dsz + d] = (half_t)y;
}

// C[row, col] = A[M,256] @ W[N,256]^T (+bias)(relu?)(+resid)(*rowscale)
// 8 waves/block: wave w -> rows blockIdx.x*128 + w*16; cols = blockIdx.y*64 .. +63
// Weight panel (64x256 f16, 32 KB) staged once per block into LDS by the
// Tensor Data Mover; B fragments then come from LDS (ds_load) while A streams
// from global with prefetch.
__global__ void __launch_bounds__(256) k_gemm(
    const half_t* __restrict__ A, const half_t* __restrict__ W,
    const float* __restrict__ bias, const float* __restrict__ resid,
    const float* __restrict__ rowscale,
    float* __restrict__ outF, half_t* __restrict__ outH, int ldo, int relu) {
  __shared__ __align__(128) half_t sW[64 * 256];   // weight panel [col][k]
  const int K = Dsz;
  int wave = threadIdx.x >> 5;
  int lane = threadIdx.x & 31;
  int row0 = blockIdx.x * 128 + wave * 16;
  int col0 = blockIdx.y * 64;

#if HAVE_TDM
  if (wave == 0) {
    tdm_load_2d_f16(W + (size_t)col0 * K, (unsigned)(size_t)(void*)sW, 64, (unsigned)K, (unsigned)K);
    __builtin_amdgcn_s_wait_tensorcnt(0);
  }
#else
  for (int idx = threadIdx.x; idx < 64 * 256 / 8; idx += 256)
    *(v8h*)&sW[idx * 8] = *(const v8h*)&W[(size_t)col0 * K + idx * 8];
#endif
  __syncthreads();

  v8f acc0 = {}, acc1 = {}, acc2 = {}, acc3 = {};
  const half_t* Arow = A + (size_t)row0 * K;
#pragma unroll
  for (int kk = 0; kk < K; kk += 32) {
    __builtin_prefetch(Arow + kk + 1024, 0, 1);   // global_prefetch_b8: next A panel
    v16h a = load_a_frag(Arow + kk, K);
    acc0 = WMMA_F32_F16(a, load_b_frag(sW + (size_t) 0 * K + kk, K), acc0);
    acc1 = WMMA_F32_F16(a, load_b_frag(sW + (size_t)16 * K + kk, K), acc1);
    acc2 = WMMA_F32_F16(a, load_b_frag(sW + (size_t)32 * K + kk, K), acc2);
    acc3 = WMMA_F32_F16(a, load_b_frag(sW + (size_t)48 * K + kk, K), acc3);
  }
  int n = lane & 15, mb = (lane >> 4) * 8;
  v8f accs[4] = {acc0, acc1, acc2, acc3};
#pragma unroll
  for (int j = 0; j < 4; j++) {
    int col = col0 + j * 16 + n;
    float bv = bias ? bias[col] : 0.f;
#pragma unroll
    for (int r = 0; r < 8; r++) {
      int row = row0 + mb + r;
      float v = accs[j][r] + bv;
      if (relu) v = fmaxf(v, 0.f);
      if (resid) v += resid[(size_t)row * ldo + col];
      if (rowscale) v *= rowscale[row];
      if (outF) outF[(size_t)row * ldo + col] = v;
      if (outH) outH[(size_t)row * ldo + col] = (half_t)v;
    }
  }
}

// Fused causal attention per (b, h): one wave per block.
__global__ void __launch_bounds__(32) k_attn(const half_t* __restrict__ qkvH,
                                             half_t* __restrict__ oH) {
  __shared__ __align__(128) half_t vt[DHd][224];  // v transposed, keys padded to 224
  __shared__ __align__(128) float  sc[16][224];   // raw scores for one 16-query block
  __shared__ __align__(128) half_t pr[16][224];   // softmax probs (f16)
  int lane = threadIdx.x;
  int bh = blockIdx.x;
  int b = bh >> 2, h = bh & 3;
  const size_t base = (size_t)b * Ssz * 768;

  for (int idx = lane; idx < DHd * 224; idx += 32) {
    int d = idx / 224, kj = idx % 224;
    half_t v = (half_t)0.f;
    if (kj < Ssz) v = qkvH[base + (size_t)kj * 768 + 512 + h * DHd + d];
    vt[d][kj] = v;
  }
  __syncthreads();

  const float scale = 0.125f;  // 1/sqrt(64)
  for (int qb = 0; qb < 13; qb++) {
    v16h a0 = load_a_q(qkvH, b, qb, h, 0);
    v16h a1 = load_a_q(qkvH, b, qb, h, 1);
    for (int kt = 0; kt < 13; kt++) {
      v8f acc = {};
      acc = WMMA_F32_F16(a0, load_b_k(qkvH, b, kt, h, 0), acc);
      acc = WMMA_F32_F16(a1, load_b_k(qkvH, b, kt, h, 1), acc);
      int n = lane & 15, mb = (lane >> 4) * 8;
#pragma unroll
      for (int r = 0; r < 8; r++) sc[mb + r][kt * 16 + n] = acc[r] * scale;
    }
    __syncthreads();

    // causal softmax, one row at a time across the wave (wave32 shuffles)
    for (int m = 0; m < 16; m++) {
      int qi = qb * 16 + m;
      float rr[7]; float mx = -3.0e38f;
#pragma unroll
      for (int j = 0; j < 7; j++) {
        int c = lane + 32 * j;
        float v = (qi < Ssz && c <= qi) ? sc[m][c] : -3.0e38f;
        rr[j] = v; mx = fmaxf(mx, v);
      }
#pragma unroll
      for (int o = 16; o > 0; o >>= 1) mx = fmaxf(mx, __shfl_xor(mx, o, 32));
      float sum = 0.f;
#pragma unroll
      for (int j = 0; j < 7; j++) {
        float e = (rr[j] > -1.0e37f) ? __expf(rr[j] - mx) : 0.f;
        rr[j] = e; sum += e;
      }
#pragma unroll
      for (int o = 16; o > 0; o >>= 1) sum += __shfl_xor(sum, o, 32);
      float inv = (qi < Ssz) ? 1.f / sum : 0.f;
#pragma unroll
      for (int j = 0; j < 7; j++) pr[m][lane + 32 * j] = (half_t)(rr[j] * inv);
    }
    __syncthreads();

    // out[16, 64] = probs[16, 224] @ v[224, 64]  (v transposed in LDS)
    v8f o0 = {}, o1 = {}, o2 = {}, o3 = {};
    const half_t* prp = &pr[0][0];
    const half_t* vtp = &vt[0][0];
    for (int ks = 0; ks < 7; ks++) {
      v16h pa = load_a_frag(prp + ks * 32, 224);
      o0 = WMMA_F32_F16(pa, load_b_frag(vtp + (size_t)0  * 224 + ks * 32, 224), o0);
      o1 = WMMA_F32_F16(pa, load_b_frag(vtp + (size_t)16 * 224 + ks * 32, 224), o1);
      o2 = WMMA_F32_F16(pa, load_b_frag(vtp + (size_t)32 * 224 + ks * 32, 224), o2);
      o3 = WMMA_F32_F16(pa, load_b_frag(vtp + (size_t)48 * 224 + ks * 32, 224), o3);
    }
    int n = lane & 15, mb = (lane >> 4) * 8;
    v8f oo[4] = {o0, o1, o2, o3};
#pragma unroll
    for (int dt = 0; dt < 4; dt++)
#pragma unroll
      for (int r = 0; r < 8; r++) {
        int qi = qb * 16 + mb + r;
        if (qi < Ssz)
          oH[(size_t)(b * Ssz + qi) * Dsz + h * DHd + dt * 16 + n] = (half_t)oo[dt][r];
      }
    __syncthreads();
  }
}

__global__ void k_final(const float* __restrict__ seqsF, const float* __restrict__ g,
                        const float* __restrict__ bb, const float* __restrict__ item_emb,
                        const int* __restrict__ pos_seqs, const int* __restrict__ neg_seqs,
                        float* __restrict__ out) {
  __shared__ float red[256];
  int t = blockIdx.x, d = threadIdx.x;
  float x = seqsF[(size_t)t * Dsz + d];
  float mean = block_reduce(x, red) * (1.0f / Dsz);
  float c = x - mean;
  float var = block_reduce(c * c, red) * (1.0f / Dsz);
  float lf = c * rsqrtf(var + EPS) * g[d] + bb[d];
  float pe = item_emb[(size_t)pos_seqs[t] * Dsz + d];
  float ne = item_emb[(size_t)neg_seqs[t] * Dsz + d];
  float ps = block_reduce(lf * pe, red);
  if (d == 0) out[t] = ps;
  float ns = block_reduce(lf * ne, red);
  if (d == 0) out[Mtok + t] = ns;
}

// ---------------------------------------------------------------------------
extern "C" void kernel_launch(void* const* d_in, const int* in_sizes, int n_in,
                              void* d_out, int out_size, void* d_ws, size_t ws_size,
                              hipStream_t stream) {
  (void)in_sizes; (void)n_in; (void)out_size; (void)ws_size;
  const float* item_emb = (const float*)d_in[0];
  const float* pos_emb  = (const float*)d_in[1];
  const float* ln1_g = (const float*)d_in[2];
  const float* ln1_b = (const float*)d_in[3];
  const float* in_w  = (const float*)d_in[4];
  const float* in_b  = (const float*)d_in[5];
  const float* out_w = (const float*)d_in[6];
  const float* out_b = (const float*)d_in[7];
  const float* ln2_g = (const float*)d_in[8];
  const float* ln2_b = (const float*)d_in[9];
  const float* c1_w  = (const float*)d_in[10];
  const float* c1_b  = (const float*)d_in[11];
  const float* c2_w  = (const float*)d_in[12];
  const float* c2_b  = (const float*)d_in[13];
  const float* last_g = (const float*)d_in[14];
  const float* last_b = (const float*)d_in[15];
  const int* log_seqs = (const int*)d_in[17];
  const int* pos_seqs = (const int*)d_in[18];
  const int* neg_seqs = (const int*)d_in[19];

  char* w = (char*)d_ws;
  size_t off = 0;
  auto alloc = [&](size_t bytes) { void* p = w + off; off += (bytes + 255) & ~(size_t)255; return p; };
  float*  seqsF = (float*) alloc((size_t)Mtok * Dsz * 4);
  float*  qF    = (float*) alloc((size_t)Mtok * Dsz * 4);
  half_t* seqsH = (half_t*)alloc((size_t)Mtok * Dsz * 2);
  half_t* qH    = (half_t*)alloc((size_t)Mtok * Dsz * 2);
  half_t* qkvH  = (half_t*)alloc((size_t)Mtok * 768 * 2);
  half_t* oH    = (half_t*)alloc((size_t)Mtok * Dsz * 2);
  half_t* h1H   = qkvH;  // FFN hidden reuses QKV space (attention already consumed)
  float*  maskF = (float*) alloc((size_t)Mtok * 4);
  half_t* inWH  = (half_t*)alloc((size_t)NBk * 768 * Dsz * 2);
  half_t* outWH = (half_t*)alloc((size_t)NBk * Dsz * Dsz * 2);
  half_t* c1WH  = (half_t*)alloc((size_t)NBk * Dsz * Dsz * 2);
  half_t* c2WH  = (half_t*)alloc((size_t)NBk * Dsz * Dsz * 2);

  int nInW = NBk * 768 * Dsz, nSq = NBk * Dsz * Dsz;
  k_cvt<<<(nInW + 255) / 256, 256, 0, stream>>>(in_w, inWH, nInW);
  k_cvt<<<(nSq + 255) / 256, 256, 0, stream>>>(out_w, outWH, nSq);
  k_cvt<<<(nSq + 255) / 256, 256, 0, stream>>>(c1_w, c1WH, nSq);
  k_cvt<<<(nSq + 255) / 256, 256, 0, stream>>>(c2_w, c2WH, nSq);

  k_embed<<<Mtok, 256, 0, stream>>>(item_emb, pos_emb, log_seqs, seqsF, seqsH, maskF);

  for (int i = 0; i < NBk; i++) {
    // Q = LN1(seqs)
    k_layernorm<<<Mtok, 256, 0, stream>>>(seqsF, ln1_g + i * Dsz, ln1_b + i * Dsz, qF, qH);
    // q = Q @ Wq^T + bq  -> qkv cols [0,256)
    k_gemm<<<dim3(Mtok / 128, 4), 256, 0, stream>>>(
        qH, inWH + (size_t)i * 768 * Dsz, in_b + i * 768,
        nullptr, nullptr, nullptr, qkvH, 768, 0);
    // k,v = seqs @ Wkv^T + bkv -> qkv cols [256,768)
    k_gemm<<<dim3(Mtok / 128, 8), 256, 0, stream>>>(
        seqsH, inWH + (size_t)i * 768 * Dsz + (size_t)256 * Dsz, in_b + i * 768 + 256,
        nullptr, nullptr, nullptr, qkvH + 256, 768, 0);
    // causal attention
    k_attn<<<Bsz * Hh, 32, 0, stream>>>(qkvH, oH);
    // seqs = Q + o @ Wo^T + bo
    k_gemm<<<dim3(Mtok / 128, 4), 256, 0, stream>>>(
        oH, outWH + (size_t)i * Dsz * Dsz, out_b + i * Dsz,
        qF, nullptr, seqsF, nullptr, 256, 0);
    // seqs = LN2(seqs)
    k_layernorm<<<Mtok, 256, 0, stream>>>(seqsF, ln2_g + i * Dsz, ln2_b + i * Dsz, seqsF, seqsH);
    // h1 = relu(seqs @ W1^T + b1)
    k_gemm<<<dim3(Mtok / 128, 4), 256, 0, stream>>>(
        seqsH, c1WH + (size_t)i * Dsz * Dsz, c1_b + i * Dsz,
        nullptr, nullptr, nullptr, h1H, 256, 1);
    // seqs = (h1 @ W2^T + b2 + seqs) * mask
    k_gemm<<<dim3(Mtok / 128, 4), 256, 0, stream>>>(
        h1H, c2WH + (size_t)i * Dsz * Dsz, c2_b + i * Dsz,
        seqsF, maskF, seqsF, seqsH, 256, 0);
  }

  k_final<<<Mtok, 256, 0, stream>>>(seqsF, last_g, last_b, item_emb,
                                    pos_seqs, neg_seqs, (float*)d_out);
}
